// GATNet_90555090469364
// MI455X (gfx1250) — matrix operand order repaired and checked by reading, not compiled, not run on hardware
//
#include <hip/hip_runtime.h>
#include <hip/hip_bf16.h>

#define NODES  50000
#define EDGES  300000
#define CH     256
#define EPRIME (EDGES + NODES)

typedef __attribute__((ext_vector_type(16))) _Float16 v16h;
typedef __attribute__((ext_vector_type(8)))  _Float16 v8h;
typedef __attribute__((ext_vector_type(8)))  float    v8f;

// ---------------- helpers ----------------

// Order-preserving map float -> u32 so unsigned atomicMax == float max.
__device__ __forceinline__ unsigned f2ord(float f) {
  unsigned b = __float_as_uint(f);
  return (b & 0x80000000u) ? ~b : (b | 0x80000000u);
}
__device__ __forceinline__ float ord2f(unsigned u) {
  return __uint_as_float((u & 0x80000000u) ? (u & 0x7FFFFFFFu) : ~u);
}

__device__ __forceinline__ void edge_sd(const int* __restrict__ ei, int e, int& s, int& d) {
  if (e < EDGES) { s = ei[e]; d = ei[EDGES + e]; }
  else           { s = e - EDGES; d = s; }           // appended self-loops
}

// ---------------- elementwise / conversion kernels ----------------

__global__ void k_cvt_f16(const float* __restrict__ in, _Float16* __restrict__ out, int n) {
  int i = blockIdx.x * blockDim.x + threadIdx.x;
  if (i < n) out[i] = (_Float16)in[i];
}

// Wt[c][k] = (f16) W[k][c]  for a 256x256 matrix (column-major B operand).
__global__ void k_cvt_f16_t256(const float* __restrict__ W, _Float16* __restrict__ Wt) {
  int i = blockIdx.x * blockDim.x + threadIdx.x;   // 65536 threads total
  int k = i >> 8, c = i & 255;
  Wt[(c << 8) + k] = (_Float16)W[i];
}

__global__ void k_fill_u32(unsigned* __restrict__ p, unsigned v, int n) {
  int i = blockIdx.x * blockDim.x + threadIdx.x;
  if (i < n) p[i] = v;
}

// xh16[n][c] = (f16) relu(acc[n][c] + b[c])   (feeds next GEMM)
__global__ void k_bias_relu_cvt(const float* __restrict__ acc, const float* __restrict__ b,
                                _Float16* __restrict__ out16, int n) {
  int i = blockIdx.x * blockDim.x + threadIdx.x;
  if (i >= n) return;
  float v = acc[i] + b[i & (CH - 1)];
  out16[i] = (_Float16)fmaxf(v, 0.0f);
}

// ---------------- WMMA GEMM: out[M][CH] = A[M][CH](f16) x B, B given column-major ----------------
// Register-blocked: each wave computes a 16x64 output slab (1 M-tile x 4 N-tiles).
// A is loaded once per K-step and reused across 4 WMMAs -> 2.5 b128 loads per WMMA.
// A : row-major f16 [M][CH]
// Bt: f16 [CH cols][CH k]  (logical B[k][c] stored at Bt[c*CH + k])
// out: f32 [M][CH], optional bias added per column.
__global__ __launch_bounds__(256)
void k_gemm_f16_wmma(const _Float16* __restrict__ A,
                     const _Float16* __restrict__ Bt,
                     const float*    __restrict__ bias,
                     float*          __restrict__ out,
                     int M) {
  const int lane = threadIdx.x & 31;
  const int wave = threadIdx.x >> 5;
  const int NT4  = (CH / 16) / 4;                  // 4 column groups of 4 tiles
  const int group = blockIdx.x * 8 + wave;         // 8 waves/block
  const int tm = group / NT4;
  const int tg = group - tm * NT4;
  if (tm * 16 >= M) return;

  const int r  = lane & 15;                        // row (A) / col (B) within tile
  const int hi = lane >> 4;                        // half-wave selects K sub-range

  const _Float16* arow = A  + (size_t)(tm * 16 + r) * CH;
  const _Float16* bcol = Bt + (size_t)(tg * 64 + r) * CH;   // tile j at +j*16*CH

  v8f c0 = {}, c1 = {}, c2 = {}, c3 = {};
  #pragma unroll
  for (int kk = 0; kk < CH; kk += 32) {
    // A k order used here (contiguous, B matches): lanes 0-15 -> K[kk..kk+15];
    //                     lanes 16-31 -> K[kk+16..kk+31]
    union { v16h v; v8h h[2]; } a;
    a.h[0] = *(const v8h*)(arow + kk + hi * 16);
    a.h[1] = *(const v8h*)(arow + kk + hi * 16 + 8);
    // B k order used here (same as A): lanes 0-15 hold K[kk..kk+15], lanes 16-31 K[kk+16..kk+31], N = lane&15
    v16h b0 = *(const v16h*)(bcol + (size_t) 0 * CH + kk + hi * 16);
    v16h b1 = *(const v16h*)(bcol + (size_t)16 * CH + kk + hi * 16);
    v16h b2 = *(const v16h*)(bcol + (size_t)32 * CH + kk + hi * 16);
    v16h b3 = *(const v16h*)(bcol + (size_t)48 * CH + kk + hi * 16);
    c0 = __builtin_amdgcn_wmma_f32_16x16x32_f16(false, a.v, false, b0, (short)0, c0, false, false);
    c1 = __builtin_amdgcn_wmma_f32_16x16x32_f16(false, a.v, false, b1, (short)0, c1, false, false);
    c2 = __builtin_amdgcn_wmma_f32_16x16x32_f16(false, a.v, false, b2, (short)0, c2, false, false);
    c3 = __builtin_amdgcn_wmma_f32_16x16x32_f16(false, a.v, false, b3, (short)0, c3, false, false);
  }

  // C/D 16x16 f32 layout: N = lane&15; VGPR v -> M = v + (lane>=16 ? 8 : 0)
  float* obase = out + (size_t)(tm * 16 + hi * 8) * CH + tg * 64 + r;
  v8f cc[4] = { c0, c1, c2, c3 };
  #pragma unroll
  for (int j = 0; j < 4; ++j) {
    const float badd = bias ? bias[tg * 64 + j * 16 + r] : 0.0f;
    #pragma unroll
    for (int v = 0; v < 8; ++v) obase[j * 16 + (size_t)v * CH] = cc[j][v] + badd;
  }
}

// ---------------- per-node attention logits: alpha = h . a ----------------
__global__ void k_alpha(const float* __restrict__ h,
                        const float* __restrict__ avs, const float* __restrict__ avd,
                        float* __restrict__ asrc, float* __restrict__ adst) {
  const int lane = threadIdx.x & 31;
  const int n = blockIdx.x * 8 + (threadIdx.x >> 5);   // wave per node
  if (n >= NODES) return;
  const float* row = h + (size_t)n * CH;
  float s0 = 0.0f, s1 = 0.0f;
  #pragma unroll
  for (int c = lane; c < CH; c += 32) {
    float v = row[c];
    s0 += v * avs[c];
    s1 += v * avd[c];
  }
  #pragma unroll
  for (int o = 16; o > 0; o >>= 1) {
    s0 += __shfl_down(s0, o, 32);
    s1 += __shfl_down(s1, o, 32);
  }
  if (lane == 0) { asrc[n] = s0; adst[n] = s1; }
}

// ---------------- edge softmax ----------------
__global__ void k_edge_max(const int* __restrict__ ei,
                           const float* __restrict__ asrc, const float* __restrict__ adst,
                           float* __restrict__ ex, unsigned* __restrict__ mkey) {
  int e = blockIdx.x * blockDim.x + threadIdx.x;
  if (e >= EPRIME) return;
  int s, d; edge_sd(ei, e, s, d);
  float v = asrc[s] + adst[d];
  v = (v > 0.0f) ? v : 0.2f * v;                  // leaky_relu(0.2)
  ex[e] = v;
  atomicMax(&mkey[d], f2ord(v));                  // segment_max over dst
}

__global__ void k_edge_exp(const int* __restrict__ ei, const unsigned* __restrict__ mkey,
                           float* __restrict__ ex, float* __restrict__ den) {
  int e = blockIdx.x * blockDim.x + threadIdx.x;
  if (e >= EPRIME) return;
  int s, d; edge_sd(ei, e, s, d);
  (void)s;
  float t = __expf(ex[e] - ord2f(mkey[d]));
  ex[e] = t;
  atomicAdd(&den[d], t);                          // segment_sum over dst
}

// ---------------- weighted scatter-aggregate: acc[d] += w * h[s] ----------------
__global__ void k_scatter(const int* __restrict__ ei, const float* __restrict__ ex,
                          const float* __restrict__ den, const float* __restrict__ h,
                          float* __restrict__ acc) {
  const int lane = threadIdx.x & 31;
  const int e = blockIdx.x * 8 + (threadIdx.x >> 5);   // wave per edge
  if (e >= EPRIME) return;
  int s, d; edge_sd(ei, e, s, d);
  const float w = ex[e] / den[d];
  const float* hs = h   + (size_t)s * CH;
  float*       ad = acc + (size_t)d * CH;
  #pragma unroll
  for (int c = lane; c < CH; c += 32) atomicAdd(&ad[c], w * hs[c]);
}

// ---------------- driver ----------------
extern "C" void kernel_launch(void* const* d_in, const int* in_sizes, int n_in,
                              void* d_out, int out_size, void* d_ws, size_t ws_size,
                              hipStream_t stream) {
  (void)in_sizes; (void)n_in; (void)out_size; (void)ws_size;

  const float* x   = (const float*)d_in[0];
  const int*   ei  = (const int*)  d_in[1];
  const float* W1  = (const float*)d_in[2];
  const float* a1s = (const float*)d_in[3];
  const float* a1d = (const float*)d_in[4];
  const float* b1  = (const float*)d_in[5];
  const float* W2  = (const float*)d_in[6];
  const float* a2s = (const float*)d_in[7];
  const float* a2d = (const float*)d_in[8];
  const float* b2  = (const float*)d_in[9];
  const float* Wl  = (const float*)d_in[10];
  const float* bl  = (const float*)d_in[11];
  float* out = (float*)d_out;

  // workspace carve-up (256B aligned)
  char* ws = (char*)d_ws;
  size_t off = 0;
  auto carve = [&](size_t bytes) -> char* {
    char* p = ws + off;
    off = (off + bytes + 255) & ~(size_t)255;
    return p;
  };
  const size_t nc = (size_t)NODES * CH;
  _Float16* xh   = (_Float16*)carve(nc * 2);            // f16 activations (reused both layers)
  _Float16* Wt1  = (_Float16*)carve((size_t)CH * CH * 2);
  _Float16* Wt2  = (_Float16*)carve((size_t)CH * CH * 2);
  _Float16* Wl16 = (_Float16*)carve((size_t)CH * CH * 2);
  float*    h    = (float*)   carve(nc * 4);            // post-GEMM features
  float*    asrc = (float*)   carve((size_t)NODES * 4);
  float*    adst = (float*)   carve((size_t)NODES * 4);
  unsigned* mkey = (unsigned*)carve((size_t)NODES * 4);
  float*    den  = (float*)   carve((size_t)NODES * 4);
  float*    ex   = (float*)   carve((size_t)EPRIME * 4);
  float*    acc  = out;   // reuse d_out as aggregation buffer; final GEMM overwrites it

  const int T = 256;
  const int NCB   = (int)((nc + T - 1) / T);            // 50000
  const int NB    = (NODES + T - 1) / T;
  const int EB    = (EPRIME + T - 1) / T;
  // 16x64 slab per wave: groups = (M/16) * 4, 8 waves/block
  const int GROUPS = (NODES / 16) * 4;                  // 12500
  const int GEMMB  = (GROUPS + 7) / 8;                  // 1563 (tail waves exit)

  // one-time f16 conversions
  k_cvt_f16     <<<NCB,           T, 0, stream>>>(x,  xh, (int)nc);
  k_cvt_f16_t256<<<CH * CH / T,   T, 0, stream>>>(W1, Wt1);
  k_cvt_f16_t256<<<CH * CH / T,   T, 0, stream>>>(W2, Wt2);
  k_cvt_f16     <<<CH * CH / T,   T, 0, stream>>>(Wl, Wl16, CH * CH);   // x@Wl.T: no transpose needed

  for (int layer = 0; layer < 2; ++layer) {
    const _Float16* Wt = layer ? Wt2 : Wt1;
    const float* avs = layer ? a2s : a1s;
    const float* avd = layer ? a2d : a1d;
    const float* bb  = layer ? b2  : b1;

    k_gemm_f16_wmma<<<GEMMB, 256, 0, stream>>>(xh, Wt, nullptr, h, NODES);   // h = x @ W
    k_alpha        <<<NODES / 8, 256, 0, stream>>>(h, avs, avd, asrc, adst);

    k_fill_u32<<<NB,  T, 0, stream>>>(mkey, 0u, NODES);          // ~ -inf sentinel in ordered-key space
    k_fill_u32<<<NB,  T, 0, stream>>>((unsigned*)den, 0u, NODES);
    k_fill_u32<<<NCB, T, 0, stream>>>((unsigned*)acc, 0u, (int)nc);

    k_edge_max<<<EB, T, 0, stream>>>(ei, asrc, adst, ex, mkey);
    k_edge_exp<<<EB, T, 0, stream>>>(ei, mkey, ex, den);
    k_scatter <<<(EPRIME + 7) / 8, 256, 0, stream>>>(ei, ex, den, h, acc);

    k_bias_relu_cvt<<<NCB, T, 0, stream>>>(acc, bb, xh, (int)nc); // relu(acc+b) -> f16 for next GEMM
  }

  // final linear: out = x @ Wl.T + bl  (Wl row-major is already the column-major B operand)
  k_gemm_f16_wmma<<<GEMMB, 256, 0, stream>>>(xh, Wl16, bl, out, NODES);
}
